// GrassMergeGen_17076789969244
// MI455X (gfx1250) — matrix operand order repaired and checked by reading, not compiled, not run on hardware
//
#include <hip/hip_runtime.h>

// ---------- types ----------
typedef __attribute__((ext_vector_type(16))) __bf16 v16bf;
typedef __attribute__((ext_vector_type(8)))  float  v8f;

union Frag16 {            // 16 bf16 = 32 bytes = 8 VGPRs
    v16bf v;
    uint4 q[2];
    unsigned u[8];
};

__device__ __forceinline__ unsigned bf16_rne(float x) {
    unsigned u = __float_as_uint(x);
    return (u + 0x7FFFu + ((u >> 16) & 1u)) >> 16;   // round-to-nearest-even
}

#define NCODES 2048
#define DDIM   128
#define NROWS  32768   // 8192 * 4 slots
#define NBATCH 8192

// ---------- pre-pass: W (f32) -> W_hi/W_lo (bf16 row-major) + ||w||^2 ----------
__global__ void __launch_bounds__(32) vq_prep(const float* __restrict__ W,
                                              unsigned short* __restrict__ whi,
                                              unsigned short* __restrict__ wlo,
                                              float* __restrict__ wsq) {
    int c    = blockIdx.x;       // code id
    int lane = threadIdx.x;      // 0..31, each lane: 4 elements
    float4 wv = ((const float4*)(W + (size_t)c * DDIM))[lane];
    float v[4] = {wv.x, wv.y, wv.z, wv.w};
    unsigned h[4], l[4];
    float s = 0.f;
#pragma unroll
    for (int i = 0; i < 4; ++i) {
        h[i] = bf16_rne(v[i]);
        float hf = __uint_as_float(h[i] << 16);
        l[i] = bf16_rne(v[i] - hf);
        s += v[i] * v[i];
    }
    uint2 hp, lp;
    hp.x = h[0] | (h[1] << 16); hp.y = h[2] | (h[3] << 16);
    lp.x = l[0] | (l[1] << 16); lp.y = l[2] | (l[3] << 16);
    *(uint2*)(whi + (size_t)c * DDIM + lane * 4) = hp;
    *(uint2*)(wlo + (size_t)c * DDIM + lane * 4) = lp;
#pragma unroll
    for (int m = 16; m >= 1; m >>= 1) s += __shfl_xor(s, m, 32);
    if (lane == 0) wsq[c] = s;
}

// ---------- helpers ----------
__device__ __forceinline__ void loadB(const unsigned short* __restrict__ bh,
                                      const unsigned short* __restrict__ bl,
                                      Frag16 (&bhi)[4], Frag16 (&blo)[4]) {
#pragma unroll
    for (int w = 0; w < 4; ++w) {
        bhi[w].q[0] = *(const uint4*)(bh + w * 32);
        bhi[w].q[1] = *(const uint4*)(bh + w * 32 + 16);
        blo[w].q[0] = *(const uint4*)(bl + w * 32);
        blo[w].q[1] = *(const uint4*)(bl + w * 32 + 16);
    }
}

__device__ __forceinline__ void procChunk(const Frag16 (&ahi)[4], const Frag16 (&alo)[4],
                                          const Frag16 (&bhi)[4], const Frag16 (&blo)[4],
                                          float wsqv, int c,
                                          float (&minv)[8], int (&mini)[8]) {
    v8f acc0 = {0.f,0.f,0.f,0.f,0.f,0.f,0.f,0.f};
    v8f acc1 = {0.f,0.f,0.f,0.f,0.f,0.f,0.f,0.f};
#pragma unroll
    for (int w = 0; w < 4; ++w) {
        acc0 = __builtin_amdgcn_wmma_f32_16x16x32_bf16(false, ahi[w].v, false, bhi[w].v,
                                                       (short)0, acc0, false, false);
        acc1 = __builtin_amdgcn_wmma_f32_16x16x32_bf16(false, ahi[w].v, false, blo[w].v,
                                                       (short)0, acc1, false, false);
        acc1 = __builtin_amdgcn_wmma_f32_16x16x32_bf16(false, alo[w].v, false, bhi[w].v,
                                                       (short)0, acc1, false, false);
    }
    // dist (up to +||f||^2 const per row) = ||w||^2 - 2 f.w
#pragma unroll
    for (int r = 0; r < 8; ++r) {
        float d = fmaf(-2.0f, acc0[r] + acc1[r], wsqv);
        bool take = d < minv[r];
        minv[r] = take ? d : minv[r];
        mini[r] = take ? c : mini[r];
    }
}

// ---------- main: argmin over codes via split-bf16 WMMA, exact rescore ----------
__global__ void __launch_bounds__(256) vq_main(const float* __restrict__ f,
                                               const float* __restrict__ W,
                                               const unsigned short* __restrict__ whi,
                                               const unsigned short* __restrict__ wlo,
                                               const float* __restrict__ wsq,
                                               float* __restrict__ loss,
                                               float* __restrict__ out) {
    const int lane = threadIdx.x & 31;
    const int wid  = threadIdx.x >> 5;
    const int gw   = blockIdx.x * 8 + wid;      // global wave id
    const int rowbase = gw * 16;                // 16 rows per wave
    const int half = lane >> 4;                 // 0: M 0-7 / K-lo group, 1: M 8-15 / K-hi group
    const int lr   = lane & 15;

    // ---- build A fragments (16 rows x 128 D, bf16 hi+lo) held for whole loop ----
    // A 16-bit 16x32 layout: lane<16 -> row=lane, elems 0..7 = K 0..7, 8..15 = K 16..23
    //                        lane>=16 -> row=lane-16, K offset +8
    Frag16 ahi[4], alo[4];
    {
        const float* fr = f + (size_t)(rowbase + lr) * DDIM;
        const int aoff = half * 8;
#pragma unroll
        for (int w = 0; w < 4; ++w) {
            const float* p = fr + w * 32 + aoff;
            float4 a0 = *(const float4*)(p);
            float4 a1 = *(const float4*)(p + 4);
            float4 b0 = *(const float4*)(p + 16);
            float4 b1 = *(const float4*)(p + 20);
            float x[16] = {a0.x,a0.y,a0.z,a0.w, a1.x,a1.y,a1.z,a1.w,
                           b0.x,b0.y,b0.z,b0.w, b1.x,b1.y,b1.z,b1.w};
#pragma unroll
            for (int pq = 0; pq < 8; ++pq) {
                float v0 = x[2*pq], v1 = x[2*pq + 1];
                unsigned h0 = bf16_rne(v0), h1 = bf16_rne(v1);
                float hf0 = __uint_as_float(h0 << 16);
                float hf1 = __uint_as_float(h1 << 16);
                unsigned l0 = bf16_rne(v0 - hf0), l1 = bf16_rne(v1 - hf1);
                ahi[w].u[pq] = h0 | (h1 << 16);
                alo[w].u[pq] = l0 | (l1 << 16);
            }
        }
    }

    // ---- stream all codes, track per-row running (min, argmin) ----
    float minv[8];
    int   mini[8];
#pragma unroll
    for (int r = 0; r < 8; ++r) { minv[r] = __builtin_inff(); mini[r] = 0; }

    const int bn   = lr;          // this lane's code column within the 16-code chunk
    const int boff = half * 8;    // B layout: lane<16 -> K 0..7,16..23, lane>=16 -> K 8..15,24..31

    // per-lane base pointers for this lane's code column
    const unsigned short* bhp = whi + (size_t)bn * DDIM + boff;
    const unsigned short* blp = wlo + (size_t)bn * DDIM + boff;

    // software pipeline: double-buffer the 16-code B chunk
    Frag16 b0h[4], b0l[4], b1h[4], b1l[4];
    loadB(bhp, blp, b0h, b0l);
    float ws0 = wsq[bn];

    for (int k0 = 0; k0 < NCODES; k0 += 32) {
        // prefetch chunk k0+16 while processing k0
        loadB(bhp + (size_t)(k0 + 16) * DDIM, blp + (size_t)(k0 + 16) * DDIM, b1h, b1l);
        float ws1 = wsq[k0 + 16 + bn];
        procChunk(ahi, alo, b0h, b0l, ws0, k0 + bn, minv, mini);

        // prefetch chunk k0+32 while processing k0+16 (uniform guard on last)
        if (k0 + 32 < NCODES) {
            loadB(bhp + (size_t)(k0 + 32) * DDIM, blp + (size_t)(k0 + 32) * DDIM, b0h, b0l);
            ws0 = wsq[k0 + 32 + bn];
        }
        procChunk(ahi, alo, b1h, b1l, ws1, k0 + 16 + bn, minv, mini);
    }

    // ---- reduce across the 16 code-columns (lanes 0..15 hold rows 0..7; 16..31 rows 8..15) ----
#pragma unroll
    for (int m = 1; m < 16; m <<= 1) {
#pragma unroll
        for (int r = 0; r < 8; ++r) {
            float ov = __shfl_xor(minv[r], m, 16);
            int   oi = __shfl_xor(mini[r], m, 16);
            if (ov < minv[r] || (ov == minv[r] && oi < mini[r])) { minv[r] = ov; mini[r] = oi; }
        }
    }

    // ---- gather W[j] (exact f32), write out, exact-rescore loss ----
    float lsum[16];
#pragma unroll
    for (int rr = 0; rr < 16; ++rr) {
        int j   = __shfl(mini[rr & 7], (rr < 8) ? 0 : 16, 32);
        int row = rowbase + rr;
        float4 wv4 = ((const float4*)(W + (size_t)j * DDIM))[lane];
        float4 fv4 = ((const float4*)(f + (size_t)row * DDIM))[lane];
        ((float4*)(out + (size_t)row * DDIM))[lane] = wv4;
        float dx = fv4.x - wv4.x, dy = fv4.y - wv4.y;
        float dz = fv4.z - wv4.z, dw = fv4.w - wv4.w;
        float p = dx*dx + dy*dy + dz*dz + dw*dw;
#pragma unroll
        for (int m = 16; m >= 1; m >>= 1) p += __shfl_xor(p, m, 32);
        lsum[rr] = p;
    }
    if (lane == 0) {
        // loss = mean over 4 slots of (1 + 0.25) * ||f - W_j||^2  => 1.25/4 = 0.3125
#pragma unroll
        for (int q = 0; q < 4; ++q)
            loss[gw * 4 + q] = 0.3125f * (lsum[4*q] + lsum[4*q+1] + lsum[4*q+2] + lsum[4*q+3]);
    }
}

extern "C" void kernel_launch(void* const* d_in, const int* in_sizes, int n_in,
                              void* d_out, int out_size, void* d_ws, size_t ws_size,
                              hipStream_t stream) {
    const float* f = (const float*)d_in[0];   // [8192, 512] f32
    const float* W = (const float*)d_in[1];   // [2048, 128] f32

    unsigned short* whi = (unsigned short*)d_ws;                       // 512 KB
    unsigned short* wlo = whi + (size_t)NCODES * DDIM;                 // 512 KB
    float* wsq = (float*)((char*)d_ws + (size_t)2 * NCODES * DDIM * 2); // 8 KB

    float* loss = (float*)d_out;              // [8192]
    float* out  = loss + NBATCH;              // [8192*512]

    vq_prep<<<NCODES, 32, 0, stream>>>(W, whi, wlo, wsq);
    vq_main<<<NROWS / 128, 256, 0, stream>>>(f, W, whi, wlo, wsq, loss, out);
}
